// UniMP_30846455120738
// MI455X (gfx1250) — compile-verified, matrix operand order checked by reading, not run on hardware
//
#include <hip/hip_runtime.h>
#include <hip/hip_bf16.h>
#include <stdint.h>

// UniMP / TransformerConv, 2 layers, for MI455X (gfx1250, wave32).
// GEMMs: v_wmma_f32_16x16x32_f16, 16x64 output strip per wave (A reused 4x),
// weight panel staged to LDS via GLOBAL_LOAD_ASYNC_TO_LDS_B128 when available,
// operands fetched as b128 vectors, 4 back-to-back WMMAs per K-step.
// Edge softmax/scatter: fp32 gathers + L2 atomics (the bandwidth-bound part).

typedef _Float16 f16;
typedef __attribute__((ext_vector_type(16))) _Float16 v16h;
typedef __attribute__((ext_vector_type(8)))  float    v8f;
typedef __attribute__((ext_vector_type(4)))  int      v4i;

#define FIN   128
#define HIDC  64      // per-head channels, both layers
#define H1    4       // heads layer 1

#if defined(__has_builtin)
#if __has_builtin(__builtin_amdgcn_global_load_async_to_lds_b128) && \
    __has_builtin(__builtin_amdgcn_s_wait_asynccnt)
#define USE_ASYNC_LDS 1
#endif
#endif

#ifdef USE_ASYNC_LDS
typedef __attribute__((address_space(1))) v4i* async_gptr_t;   // global src
typedef __attribute__((address_space(3))) v4i* async_lptr_t;   // LDS dst
#endif

// ---- monotonic float<->ordered-uint mapping for atomicMax on floats ----
__device__ __forceinline__ unsigned f2ord(float f) {
  unsigned u = __float_as_uint(f);
  return (u & 0x80000000u) ? ~u : (u | 0x80000000u);
}
__device__ __forceinline__ float ord2f(unsigned u) {
  unsigned v = (u & 0x80000000u) ? (u & 0x7fffffffu) : ~u;
  return __uint_as_float(v);
}

// ---------------- elementwise converts ----------------
__global__ void k_f32_to_f16(const float* __restrict__ in, f16* __restrict__ out, long n) {
  long i = (long)blockIdx.x * blockDim.x + threadIdx.x;
  long s = (long)gridDim.x * blockDim.x;
  for (; i < n; i += s) out[i] = (f16)in[i];
}

__global__ void k_relu_f32_to_f16(const float* __restrict__ in, f16* __restrict__ out, long n) {
  long i = (long)blockIdx.x * blockDim.x + threadIdx.x;
  long s = (long)gridDim.x * blockDim.x;
  for (; i < n; i += s) out[i] = (f16)fmaxf(in[i], 0.0f);
}

// W[k, n] (row-major K x Nc, fp32) -> Wt[n, k] (row-major Nc x K, fp16)
__global__ void k_convert_w_t(const float* __restrict__ W, f16* __restrict__ Wt, int K, int Nc) {
  long total = (long)K * Nc;
  long i = (long)blockIdx.x * blockDim.x + threadIdx.x;
  long s = (long)gridDim.x * blockDim.x;
  for (; i < total; i += s) {
    int k = (int)(i / Nc), n = (int)(i % Nc);
    Wt[(long)n * K + k] = (f16)W[i];
  }
}

__global__ void k_zero_u32(unsigned* __restrict__ p, long n) {
  long i = (long)blockIdx.x * blockDim.x + threadIdx.x;
  long s = (long)gridDim.x * blockDim.x;
  for (; i < n; i += s) p[i] = 0u;
}

// ---------------- WMMA GEMM: C[M,Nc] = A_f16[M,K] @ Bt_f16[Nc,K]^T + bias ----------------
// Block = 256 threads = 8 waves. Block tile: 128 rows (wave = 16 rows) x 64 cols.
// The 64-col weight panel (64*K halfs, contiguous in Bt) is staged once into LDS
// (async-to-LDS), then every wave runs 4 WMMAs per K-step reusing its A operand.
// Operand lane layout (ISA 7.12.2, 16-bit 16x32): lane = {khalf, r[3:0]};
//   per lane the 16 halfs are two contiguous 16B chunks at dword offsets
//   khalf*4 and 8+khalf*4 within the 32-half K-slice -> b128 loads.
__global__ void __launch_bounds__(256)
k_gemm_wmma(const f16* __restrict__ A, const f16* __restrict__ Bt,
            const float* __restrict__ bias, float* __restrict__ C,
            int M, int Nc, int K) {
  __shared__ __align__(16) f16 bpanel[64 * 256];   // up to K=256 -> 32 KB

  const int tid   = threadIdx.x;
  const int wave  = tid >> 5;
  const int lane  = tid & 31;
  const int n0    = blockIdx.x * 64;
  const int tm    = blockIdx.y * 8 + wave;         // 16-row tile index
  const bool valid = (tm < (M >> 4));              // whole-wave uniform

  // ---- stage 64 x K weight panel (contiguous in Bt) into LDS ----
  {
    const char* gsrc = (const char*)(Bt + (size_t)n0 * K);
    const int chunks = 8 * K;                      // 16-byte chunks: 64*K*2/16
    for (int i = tid; i < chunks; i += 256) {
#ifdef USE_ASYNC_LDS
      __builtin_amdgcn_global_load_async_to_lds_b128(
          (async_gptr_t)(gsrc + (size_t)i * 16),
          (async_lptr_t)&bpanel[i * 8],
          0, 0);
#else
      *(float4*)&bpanel[i * 8] = *(const float4*)(gsrc + (size_t)i * 16);
#endif
    }
#ifdef USE_ASYNC_LDS
    __builtin_amdgcn_s_wait_asynccnt(0);
#endif
    __syncthreads();
  }

  if (!valid) return;

  const int r     = lane & 15;                     // A row / B col within subtile
  const int khalf = lane >> 4;

  const v4i* a4  = (const v4i*)(A + (size_t)(tm * 16 + r) * K);
  const v4i* b4_0 = (const v4i*)(bpanel + (size_t)(0 * 16 + r) * K);
  const v4i* b4_1 = (const v4i*)(bpanel + (size_t)(1 * 16 + r) * K);
  const v4i* b4_2 = (const v4i*)(bpanel + (size_t)(2 * 16 + r) * K);
  const v4i* b4_3 = (const v4i*)(bpanel + (size_t)(3 * 16 + r) * K);

  union U { v16h h; v4i v[2]; };

  v8f acc0 = {}, acc1 = {}, acc2 = {}, acc3 = {};
  for (int k0 = 0; k0 < K; k0 += 32) {
    const int ci = (k0 >> 3) + khalf;              // 16B-chunk index within row
    U Aop, B0, B1, B2, B3;
    Aop.v[0] = a4[ci];    Aop.v[1] = a4[ci + 2];
    B0.v[0]  = b4_0[ci];  B0.v[1]  = b4_0[ci + 2];
    B1.v[0]  = b4_1[ci];  B1.v[1]  = b4_1[ci + 2];
    B2.v[0]  = b4_2[ci];  B2.v[1]  = b4_2[ci + 2];
    B3.v[0]  = b4_3[ci];  B3.v[1]  = b4_3[ci + 2];
    acc0 = __builtin_amdgcn_wmma_f32_16x16x32_f16(false, Aop.h, false, B0.h,
                                                  (short)0, acc0, false, false);
    acc1 = __builtin_amdgcn_wmma_f32_16x16x32_f16(false, Aop.h, false, B1.h,
                                                  (short)0, acc1, false, false);
    acc2 = __builtin_amdgcn_wmma_f32_16x16x32_f16(false, Aop.h, false, B2.h,
                                                  (short)0, acc2, false, false);
    acc3 = __builtin_amdgcn_wmma_f32_16x16x32_f16(false, Aop.h, false, B3.h,
                                                  (short)0, acc3, false, false);
  }

  // C/D layout: lane n = lane&15, M offset = (lane>>4)*8
  const int n    = lane & 15;
  const int moff = (lane >> 4) * 8;
#pragma unroll
  for (int t = 0; t < 4; ++t) {
    v8f acc = (t == 0) ? acc0 : (t == 1) ? acc1 : (t == 2) ? acc2 : acc3;
    const int col = n0 + t * 16 + n;
    const float bv = bias[col];
#pragma unroll
    for (int rr = 0; rr < 8; ++rr)
      C[(size_t)(tm * 16 + moff + rr) * Nc + col] = acc[rr] + bv;
  }
}

// ---------------- edge phase ----------------
// logits[e*H+h] = (q[dst] . k[src]) * scale ; segment max via ordered-uint atomicMax
__global__ void k_edge_logits(const float* __restrict__ q, const float* __restrict__ kk,
                              const int* __restrict__ src, const int* __restrict__ dst,
                              float* __restrict__ logits, unsigned* __restrict__ mx,
                              long EH, int H, float scale) {
  long i = (long)blockIdx.x * blockDim.x + threadIdx.x;
  long s = (long)gridDim.x * blockDim.x;
  for (; i < EH; i += s) {
    long e = i / H; int h = (int)(i % H);
    int d = dst[e], sn = src[e];
    long ip = i + s;                 // prefetch next grid-stride iteration's rows
    if (ip < EH) {
      long e2 = ip / H; int h2 = (int)(ip % H);
      __builtin_prefetch(q  + ((size_t)dst[e2] * H + h2) * HIDC, 0, 0);
      __builtin_prefetch(kk + ((size_t)src[e2] * H + h2) * HIDC, 0, 0);
    }
    const float4* qp = (const float4*)(q  + ((size_t)d  * H + h) * HIDC);
    const float4* kp = (const float4*)(kk + ((size_t)sn * H + h) * HIDC);
    float acc = 0.0f;
#pragma unroll
    for (int j = 0; j < HIDC / 4; ++j) {
      float4 a = qp[j], b = kp[j];
      acc += a.x * b.x + a.y * b.y + a.z * b.z + a.w * b.w;
    }
    float l = acc * scale;
    logits[i] = l;
    atomicMax(mx + (size_t)d * H + h, f2ord(l));
  }
}

// alpha[i] = exp(logit - segmax) (stored over logits); denom[dst,h] += alpha
__global__ void k_edge_exp(float* __restrict__ logits, const unsigned* __restrict__ mx,
                           float* __restrict__ dn, const int* __restrict__ dst,
                           long EH, int H) {
  long i = (long)blockIdx.x * blockDim.x + threadIdx.x;
  long s = (long)gridDim.x * blockDim.x;
  for (; i < EH; i += s) {
    long e = i / H; int h = (int)(i % H);
    int d = dst[e];
    float m  = ord2f(mx[(size_t)d * H + h]);
    float el = expf(logits[i] - m);
    logits[i] = el;
    atomicAdd(dn + (size_t)d * H + h, el);
  }
}

// out[dst, h, c] += (alpha/denom) * v[src, h, c] ; 64 consecutive threads per (e,h) -> coalesced
__global__ void k_edge_scatter(const float* __restrict__ alpha, const float* __restrict__ dn,
                               const float* __restrict__ v, const int* __restrict__ src,
                               const int* __restrict__ dst, float* __restrict__ out,
                               long EH, int H) {
  int  gpb = blockDim.x / HIDC;
  long g   = (long)blockIdx.x * gpb + threadIdx.x / HIDC;
  int  c   = threadIdx.x % HIDC;
  long gs  = (long)gridDim.x * gpb;
  for (; g < EH; g += gs) {
    long e = g / H; int h = (int)(g % H);
    int d = dst[e], sn = src[e];
    float a = alpha[g] / dn[(size_t)d * H + h];
    atomicAdd(out + ((size_t)d * H + h) * HIDC + c,
              a * v[((size_t)sn * H + h) * HIDC + c]);
  }
}

// ---------------- log_softmax over rows of 64, one wave32 per row ----------------
__global__ void k_log_softmax64(float* __restrict__ y, int N) {
  int row  = blockIdx.x * (blockDim.x >> 5) + (threadIdx.x >> 5);
  int lane = threadIdx.x & 31;
  if (row >= N) return;
  float* p = y + (size_t)row * 64;
  float a = p[lane], b = p[lane + 32];
  float m = fmaxf(a, b);
#pragma unroll
  for (int off = 16; off; off >>= 1) m = fmaxf(m, __shfl_xor(m, off, 32));
  float s = expf(a - m) + expf(b - m);
#pragma unroll
  for (int off = 16; off; off >>= 1) s += __shfl_xor(s, off, 32);
  float lse = logf(s) + m;
  p[lane] = a - lse;
  p[lane + 32] = b - lse;
}

// ---------------- host side ----------------
static inline unsigned nblk(long n, int bs) {
  long b = (n + bs - 1) / bs;
  return (unsigned)b;
}

extern "C" void kernel_launch(void* const* d_in, const int* in_sizes, int n_in,
                              void* d_out, int out_size, void* d_ws, size_t ws_size,
                              hipStream_t stream) {
  const float* x  = (const float*)d_in[0];
  const int*   ei = (const int*)d_in[1];
  long N = in_sizes[0] / FIN;
  long E = in_sizes[1] / 2;
  const int* src = ei;
  const int* dst = ei + E;

  const float* Wq1 = (const float*)d_in[2];  const float* bq1 = (const float*)d_in[3];
  const float* Wk1 = (const float*)d_in[4];  const float* bk1 = (const float*)d_in[5];
  const float* Wv1 = (const float*)d_in[6];  const float* bv1 = (const float*)d_in[7];
  const float* Ws1 = (const float*)d_in[8];  const float* bs1 = (const float*)d_in[9];
  const float* Wq2 = (const float*)d_in[10]; const float* bq2 = (const float*)d_in[11];
  const float* Wk2 = (const float*)d_in[12]; const float* bk2 = (const float*)d_in[13];
  const float* Wv2 = (const float*)d_in[14]; const float* bv2 = (const float*)d_in[15];
  const float* Ws2 = (const float*)d_in[16]; const float* bs2 = (const float*)d_in[17];

  const int HC1 = H1 * HIDC;   // 256

  char* ws = (char*)d_ws;
  size_t off = 0;
  auto alloc = [&](size_t bytes) -> char* {
    char* p = ws + off;
    off = (off + bytes + 255) & ~(size_t)255;
    return p;
  };
  f16*      xh    = (f16*)     alloc((size_t)N * FIN * sizeof(f16));
  f16*      h1h   = (f16*)     alloc((size_t)N * HC1 * sizeof(f16));
  float*    q1    = (float*)   alloc((size_t)N * HC1 * sizeof(float));
  float*    k1    = (float*)   alloc((size_t)N * HC1 * sizeof(float));
  float*    v1    = (float*)   alloc((size_t)N * HC1 * sizeof(float));
  float*    out1  = (float*)   alloc((size_t)N * HC1 * sizeof(float));
  float*    alpha = (float*)   alloc((size_t)E * H1 * sizeof(float));
  unsigned* mx    = (unsigned*)alloc((size_t)N * H1 * sizeof(unsigned));
  float*    dn    = (float*)   alloc((size_t)N * H1 * sizeof(float));
  f16* wtq1 = (f16*)alloc((size_t)HC1 * FIN * sizeof(f16));
  f16* wtk1 = (f16*)alloc((size_t)HC1 * FIN * sizeof(f16));
  f16* wtv1 = (f16*)alloc((size_t)HC1 * FIN * sizeof(f16));
  f16* wts1 = (f16*)alloc((size_t)HC1 * FIN * sizeof(f16));
  f16* wtq2 = (f16*)alloc((size_t)HIDC * HC1 * sizeof(f16));
  f16* wtk2 = (f16*)alloc((size_t)HIDC * HC1 * sizeof(f16));
  f16* wtv2 = (f16*)alloc((size_t)HIDC * HC1 * sizeof(f16));
  f16* wts2 = (f16*)alloc((size_t)HIDC * HC1 * sizeof(f16));
  (void)ws_size; (void)n_in;

  // layer-2 buffers alias layer-1 scratch (strictly smaller)
  float*    q2  = q1;  float* k2 = k1;  float* v2 = v1;
  float*    al2 = alpha;
  unsigned* mx2 = mx;  float* dn2 = dn;
  float*    out2 = (float*)d_out;       // N x 64, fully written by skip GEMM

  const int EB = 256;
  const float scale = 0.125f;           // 1/sqrt(64)

  // ===== layer 1 =====
  k_f32_to_f16<<<nblk(N * FIN, EB), EB, 0, stream>>>(x, xh, N * FIN);
  k_convert_w_t<<<nblk((long)FIN * HC1, EB), EB, 0, stream>>>(Wq1, wtq1, FIN, HC1);
  k_convert_w_t<<<nblk((long)FIN * HC1, EB), EB, 0, stream>>>(Wk1, wtk1, FIN, HC1);
  k_convert_w_t<<<nblk((long)FIN * HC1, EB), EB, 0, stream>>>(Wv1, wtv1, FIN, HC1);
  k_convert_w_t<<<nblk((long)FIN * HC1, EB), EB, 0, stream>>>(Ws1, wts1, FIN, HC1);

  {
    dim3 g(HC1 / 64, nblk(N >> 4, 8));  // x: 64-col panels, y: 128-row blocks
    k_gemm_wmma<<<g, 256, 0, stream>>>(xh, wtq1, bq1, q1,   (int)N, HC1, FIN);
    k_gemm_wmma<<<g, 256, 0, stream>>>(xh, wtk1, bk1, k1,   (int)N, HC1, FIN);
    k_gemm_wmma<<<g, 256, 0, stream>>>(xh, wtv1, bv1, v1,   (int)N, HC1, FIN);
    k_gemm_wmma<<<g, 256, 0, stream>>>(xh, wts1, bs1, out1, (int)N, HC1, FIN); // skip accumulator
  }

  k_zero_u32<<<nblk(N * H1, EB), EB, 0, stream>>>(mx, N * H1);
  k_zero_u32<<<nblk(N * H1, EB), EB, 0, stream>>>((unsigned*)dn, N * H1);

  long EH1 = E * H1;
  k_edge_logits<<<nblk(EH1, EB), EB, 0, stream>>>(q1, k1, src, dst, alpha, mx, EH1, H1, scale);
  k_edge_exp   <<<nblk(EH1, EB), EB, 0, stream>>>(alpha, mx, dn, dst, EH1, H1);
  k_edge_scatter<<<nblk(EH1, EB / HIDC), EB, 0, stream>>>(alpha, dn, v1, src, dst, out1, EH1, H1);

  k_relu_f32_to_f16<<<nblk(N * HC1, EB), EB, 0, stream>>>(out1, h1h, N * HC1);

  // ===== layer 2 =====
  k_convert_w_t<<<nblk((long)HC1 * HIDC, EB), EB, 0, stream>>>(Wq2, wtq2, HC1, HIDC);
  k_convert_w_t<<<nblk((long)HC1 * HIDC, EB), EB, 0, stream>>>(Wk2, wtk2, HC1, HIDC);
  k_convert_w_t<<<nblk((long)HC1 * HIDC, EB), EB, 0, stream>>>(Wv2, wtv2, HC1, HIDC);
  k_convert_w_t<<<nblk((long)HC1 * HIDC, EB), EB, 0, stream>>>(Ws2, wts2, HC1, HIDC);

  {
    dim3 g(HIDC / 64, nblk(N >> 4, 8));
    k_gemm_wmma<<<g, 256, 0, stream>>>(h1h, wtq2, bq2, q2,   (int)N, HIDC, HC1);
    k_gemm_wmma<<<g, 256, 0, stream>>>(h1h, wtk2, bk2, k2,   (int)N, HIDC, HC1);
    k_gemm_wmma<<<g, 256, 0, stream>>>(h1h, wtv2, bv2, v2,   (int)N, HIDC, HC1);
    k_gemm_wmma<<<g, 256, 0, stream>>>(h1h, wts2, bs2, out2, (int)N, HIDC, HC1); // writes all of d_out
  }

  k_zero_u32<<<nblk(N, EB), EB, 0, stream>>>(mx2, N);
  k_zero_u32<<<nblk(N, EB), EB, 0, stream>>>((unsigned*)dn2, N);

  k_edge_logits<<<nblk(E, EB), EB, 0, stream>>>(q2, k2, src, dst, al2, mx2, E, 1, scale);
  k_edge_exp   <<<nblk(E, EB), EB, 0, stream>>>(al2, mx2, dn2, dst, E, 1);
  k_edge_scatter<<<nblk(E, EB / HIDC), EB, 0, stream>>>(al2, dn2, v2, src, dst, out2, E, 1);

  // log_softmax rows of 64, one wave per row, 8 waves per block
  k_log_softmax64<<<nblk(N, 8), 256, 0, stream>>>(out2, (int)N);
}
